// MultiHeadAttention_5703716569266
// MI455X (gfx1250) — matrix operand order repaired and checked by reading, not compiled, not run on hardware
//
#include <hip/hip_runtime.h>
#include <hip/hip_bf16.h>

// Problem constants (match reference)
constexpr int CB = 8;     // batch
constexpr int CS = 1024;  // seq len
constexpr int CD = 768;   // hidden
constexpr int CH = 12;    // heads
constexpr int CE = 64;    // head dim

typedef __attribute__((ext_vector_type(8)))  __bf16 v8bf;
typedef __attribute__((ext_vector_type(16))) __bf16 v16bf;
typedef __attribute__((ext_vector_type(8)))  float  v8f;

#define V8F_ZERO {0.f,0.f,0.f,0.f,0.f,0.f,0.f,0.f}

#define HAS_ASYNC_LDS __has_builtin(__builtin_amdgcn_global_load_async_to_lds_b128)
#if HAS_ASYNC_LDS
typedef int v4i __attribute__((vector_size(16)));
typedef __attribute__((address_space(1))) v4i gv4i;   // global int4
typedef __attribute__((address_space(3))) v4i lv4i;   // LDS int4
#endif

// fp32 -> bf16 via native conversion (compiler emits v_cvt_*_bf16_f32, packs pairs)
__device__ __forceinline__ __bf16 f2bf(float f) { return (__bf16)f; }

// A-matrix fragment (16x32 bf16, MxK). Contiguous-K layout (same as frag_b):
// lane L: row M = L%16; elems 0-15 = K[kh..kh+15], kh = 16*(L>=16)
__device__ __forceinline__ v16bf frag_a(const __bf16* p, int stride, int lane) {
    int r  = lane & 15;
    int kh = lane & 16;                   // 0 or 16
    const __bf16* row = p + r * stride + kh;
    v8bf lo = *(const v8bf*)(row);        // 16B load
    v8bf hi = *(const v8bf*)(row + 8);    // 16B load
    return __builtin_shufflevector(lo, hi, 0,1,2,3,4,5,6,7,8,9,10,11,12,13,14,15);
}

// B-matrix fragment (32x16 bf16, KxN), stored with per-column K contiguous.
// lane L: col N = L%16; elems 0-15 = K[kh..kh+15], kh = 16*(L>=16)  -> one 32B load
__device__ __forceinline__ v16bf frag_b(const __bf16* p, int stride, int lane) {
    int n  = lane & 15;
    int kh = lane & 16;                   // 0 or 16
    return *(const v16bf*)(p + (size_t)n * stride + kh);
}

__device__ __forceinline__ v8f wmma_bf16(v16bf a, v16bf b, v8f c) {
    return __builtin_amdgcn_wmma_f32_16x16x32_bf16(false, a, false, b, (short)0, c, false, false);
}

__device__ __forceinline__ v8bf pack8(float4 f0, float4 f1) {
    v8bf p;
    p[0]=f2bf(f0.x); p[1]=f2bf(f0.y); p[2]=f2bf(f0.z); p[3]=f2bf(f0.w);
    p[4]=f2bf(f1.x); p[5]=f2bf(f1.y); p[6]=f2bf(f1.z); p[7]=f2bf(f1.w);
    return p;
}

// ---------------------------------------------------------------------------
// Kernel 1: fused per-head QKV projection.  grid = (CS/64, CB*CH, 3)
// Q,K stored [B,H,S,E] bf16;  V stored TRANSPOSED [B,H,E,S] bf16 so the ctx
// kernel can load its B fragments straight from global with contiguous 32B.
// ---------------------------------------------------------------------------
__global__ __launch_bounds__(256) void qkv_proj_kernel(
    const float* __restrict__ x,
    const float* __restrict__ Wq, const float* __restrict__ bq,
    const float* __restrict__ Wk, const float* __restrict__ bk,
    const float* __restrict__ Wv, const float* __restrict__ bv,
    __bf16* __restrict__ qb, __bf16* __restrict__ kb, __bf16* __restrict__ vb)
{
    const int mt    = blockIdx.x;           // 64-row tile of S
    const int bh    = blockIdx.y;
    const int which = blockIdx.z;           // 0=Q 1=K 2=V
    const int b = bh / CH, h = bh % CH;
    const float* W  = (which == 0) ? Wq : (which == 1) ? Wk : Wv;   // [H,D,E]
    const float* bi = (which == 0) ? bq : (which == 1) ? bk : bv;   // [H,E]
    __bf16* outp    = (which == 0) ? qb : (which == 1) ? kb : vb;

    __shared__ __bf16 Xs[64 * 32];   // [row][k]
    __shared__ __bf16 Ws[64 * 32];   // [n][k]  (W transposed: per-column K contiguous)

    const int tid = threadIdx.x;
    const int wave = tid >> 5, lane = tid & 31;
    const int rt = wave & 3;                 // row sub-tile (16 rows)
    const int cg = wave >> 2;                // col group: tiles {2cg, 2cg+1}

    v8f c0 = V8F_ZERO, c1 = V8F_ZERO;

    const float* xrow = x + ((size_t)b * CS + (size_t)mt * 64) * CD;

    for (int kk = 0; kk < CD; kk += 32) {
        {   // stage X tile [64 x 32] fp32 -> bf16, one b128 LDS store
            int r = tid >> 2, ks = (tid & 3) * 8;
            const float4* src = (const float4*)(xrow + (size_t)r * CD + kk + ks);
            if (kk + 32 < CD) __builtin_prefetch(xrow + (size_t)r * CD + kk + 32 + ks, 0, 0);
            *(v8bf*)&Xs[r * 32 + ks] = pack8(src[0], src[1]);
        }
        {   // stage W tile transposed -> Ws[n][k]; strided coalesced reads,
            // contiguous b128 LDS write
            int n = tid & 63, kg = tid >> 6;     // kg = 0..3 (8 k each)
            const float* src = W + ((size_t)h * CD + kk + kg * 8) * CE + n;
            v8bf pk;
            #pragma unroll
            for (int i = 0; i < 8; ++i) pk[i] = f2bf(src[(size_t)i * CE]);
            *(v8bf*)&Ws[n * 32 + kg * 8] = pk;
        }
        __syncthreads();
        v16bf a  = frag_a(&Xs[rt * 16 * 32], 32, lane);
        v16bf b0 = frag_b(&Ws[(cg * 2 + 0) * 16 * 32], 32, lane);
        v16bf b1 = frag_b(&Ws[(cg * 2 + 1) * 16 * 32], 32, lane);
        c0 = wmma_bf16(a, b0, c0);
        c1 = wmma_bf16(a, b1, c1);
        __syncthreads();
    }

    // epilogue: +bias, store bf16
    const int col   = lane & 15;
    const int rbase = mt * 64 + rt * 16 + ((lane & 16) >> 1);   // +8 for upper half-wave
    const int e0 = cg * 32 + col, e1 = e0 + 16;
    const float bias0 = bi[h * CE + e0], bias1 = bi[h * CE + e1];
    if (which == 2) {
        // V: transposed [B,H,E,S]; each lane's 8 rows are contiguous in S
        __bf16* obase = outp + ((size_t)bh * CE) * CS;
        v8bf p0, p1;
        #pragma unroll
        for (int r = 0; r < 8; ++r) { p0[r] = f2bf(c0[r] + bias0); p1[r] = f2bf(c1[r] + bias1); }
        *(v8bf*)(obase + (size_t)e0 * CS + rbase) = p0;
        *(v8bf*)(obase + (size_t)e1 * CS + rbase) = p1;
    } else {
        // Q,K: row-major [B,H,S,E]
        __bf16* obase = outp + ((size_t)bh * CS) * CE;
        #pragma unroll
        for (int r = 0; r < 8; ++r) {
            int row = rbase + r;
            obase[(size_t)row * CE + e0] = f2bf(c0[r] + bias0);
            obase[(size_t)row * CE + e1] = f2bf(c1[r] + bias1);
        }
    }
}

// ---------------------------------------------------------------------------
// Kernel 2: scores = q @ k^T / 8.   grid = (CS/64, CS/64, CB*CH)
// Fragments loaded straight from global bf16 (rows already K-contiguous).
// ---------------------------------------------------------------------------
__global__ __launch_bounds__(256) void scores_kernel(
    const __bf16* __restrict__ qb, const __bf16* __restrict__ kb,
    float* __restrict__ probs)
{
    const int qt = blockIdx.x, kt = blockIdx.y, bh = blockIdx.z;
    const int tid = threadIdx.x, wave = tid >> 5, lane = tid & 31;
    const int rt = wave & 3, cg = wave >> 2;

    const __bf16* qbase = qb + ((size_t)bh * CS + qt * 64 + rt * 16) * CE;
    const __bf16* kbase = kb + ((size_t)bh * CS + kt * 64 + cg * 32) * CE;

    v8f c0 = V8F_ZERO, c1 = V8F_ZERO;
    #pragma unroll
    for (int kk = 0; kk < CE; kk += 32) {
        v16bf a  = frag_a(qbase + kk, CE, lane);
        v16bf b0 = frag_b(kbase + kk, CE, lane);                 // k rows = B columns
        v16bf b1 = frag_b(kbase + (size_t)16 * CE + kk, CE, lane);
        c0 = wmma_bf16(a, b0, c0);
        c1 = wmma_bf16(a, b1, c1);
    }

    const int col   = lane & 15;
    const int rbase = qt * 64 + rt * 16 + ((lane & 16) >> 1);
    float* pbase = probs + (size_t)bh * CS * CS;
    const int t0 = kt * 64 + cg * 32 + col, t1 = t0 + 16;
    #pragma unroll
    for (int r = 0; r < 8; ++r) {
        int srow = rbase + r;
        pbase[(size_t)srow * CS + t0] = c0[r] * 0.125f;
        pbase[(size_t)srow * CS + t1] = c1[r] * 0.125f;
    }
}

// ---------------------------------------------------------------------------
// Kernel 3: row softmax in-place on probs (the HBM-bound stage: ~800 MB).
// One wave32 per row of 1024; float4 vectorized, single read + single write.
// grid = (CB*CH*CS/8), block 256 (8 waves)
// ---------------------------------------------------------------------------
__global__ __launch_bounds__(256) void softmax_kernel(float* __restrict__ probs)
{
    const int tid = threadIdx.x, wave = tid >> 5, lane = tid & 31;
    const size_t row = (size_t)blockIdx.x * 8 + wave;
    float4* p = (float4*)(probs + row * CS);

    float4 vals[8];
    float m = -3.4e38f;
    #pragma unroll
    for (int i = 0; i < 8; ++i) {
        vals[i] = p[i * 32 + lane];
        m = fmaxf(m, fmaxf(fmaxf(vals[i].x, vals[i].y), fmaxf(vals[i].z, vals[i].w)));
    }
    #pragma unroll
    for (int off = 16; off > 0; off >>= 1) m = fmaxf(m, __shfl_xor(m, off, 32));

    float s = 0.f;
    #pragma unroll
    for (int i = 0; i < 8; ++i) {
        vals[i].x = __expf(vals[i].x - m); vals[i].y = __expf(vals[i].y - m);
        vals[i].z = __expf(vals[i].z - m); vals[i].w = __expf(vals[i].w - m);
        s += (vals[i].x + vals[i].y) + (vals[i].z + vals[i].w);
    }
    #pragma unroll
    for (int off = 16; off > 0; off >>= 1) s += __shfl_xor(s, off, 32);

    const float inv = 1.0f / s;
    #pragma unroll
    for (int i = 0; i < 8; ++i) {
        vals[i].x *= inv; vals[i].y *= inv; vals[i].z *= inv; vals[i].w *= inv;
        p[i * 32 + lane] = vals[i];
    }
}

// ---------------------------------------------------------------------------
// Kernel 4: ctx = probs @ v.  grid = (CS/64, CB*CH)
// v is pre-transposed [B,H,E,S] -> B fragments load directly from global
// (32B contiguous per lane), no LDS staging for V at all.
// ctx stored head-concatenated bf16 [B,S,H*E] in ws.
// ---------------------------------------------------------------------------
__global__ __launch_bounds__(256) void ctx_kernel(
    const float* __restrict__ probs, const __bf16* __restrict__ vb,
    __bf16* __restrict__ ctx)
{
    const int st = blockIdx.x, bh = blockIdx.y;
    const int b = bh / CH, h = bh % CH;
    const int tid = threadIdx.x, wave = tid >> 5, lane = tid & 31;
    const int rt = wave & 3, cg = wave >> 2;

    __shared__ __bf16 Ps[64 * 32];   // [row][t]

    const float*  prow   = probs + ((size_t)bh * CS + (size_t)st * 64) * CS;
    const __bf16* vtbase = vb + ((size_t)bh * CE) * CS;   // [E][S]

    v8f c0 = V8F_ZERO, c1 = V8F_ZERO;

    for (int tt = 0; tt < CS; tt += 32) {
        {   // stage probs tile fp32 -> bf16, one b128 LDS store
            int r = tid >> 2, ts = (tid & 3) * 8;
            const float4* src = (const float4*)(prow + (size_t)r * CS + tt + ts);
            if (tt + 32 < CS) __builtin_prefetch(prow + (size_t)r * CS + tt + 32 + ts, 0, 0);
            *(v8bf*)&Ps[r * 32 + ts] = pack8(src[0], src[1]);
        }
        __syncthreads();
        v16bf a  = frag_a(&Ps[rt * 16 * 32], 32, lane);
        v16bf b0 = frag_b(vtbase + (size_t)(cg * 32) * CS + tt, CS, lane);
        v16bf b1 = frag_b(vtbase + (size_t)(cg * 32 + 16) * CS + tt, CS, lane);
        c0 = wmma_bf16(a, b0, c0);
        c1 = wmma_bf16(a, b1, c1);
        __syncthreads();
    }

    const int col   = lane & 15;
    const int rbase = st * 64 + rt * 16 + ((lane & 16) >> 1);
    const int e0 = cg * 32 + col, e1 = e0 + 16;
    #pragma unroll
    for (int r = 0; r < 8; ++r) {
        int s = rbase + r;
        ctx[((size_t)b * CS + s) * CD + h * CE + e0] = f2bf(c0[r]);
        ctx[((size_t)b * CS + s) * CD + h * CE + e1] = f2bf(c1[r]);
    }
}

// ---------------------------------------------------------------------------
// Kernel 5: out = ctx @ Wo^T + bo  (fp32 out).  grid = (CB*CS/64, CD/64)
// ctx tile is a pure bf16 copy -> use gfx1250 async global->LDS loads
// (ASYNCcnt path) when the builtin is available.
// ---------------------------------------------------------------------------
__global__ __launch_bounds__(256) void outproj_kernel(
    const __bf16* __restrict__ ctx, const float* __restrict__ Wo,
    const float* __restrict__ bo, float* __restrict__ out)
{
    const int mt = blockIdx.x;   // 64-row tile of B*S
    const int nt = blockIdx.y;   // 64-col tile of D
    const int tid = threadIdx.x, wave = tid >> 5, lane = tid & 31;
    const int rt = wave & 3, cg = wave >> 2;

    __shared__ __bf16 Xs[64 * 32];   // ctx tile [row][k]
    __shared__ __bf16 Ws[64 * 32];   // Wo tile [n][k]

    const __bf16* xbase = ctx + (size_t)mt * 64 * CD;

    v8f c0 = V8F_ZERO, c1 = V8F_ZERO;

    for (int kk = 0; kk < CD; kk += 32) {
        {   // ctx already bf16: async copy straight into LDS (16B per lane)
            int r = tid >> 2, ks = (tid & 3) * 8;
            const __bf16* gsrc = xbase + (size_t)r * CD + kk + ks;
#if HAS_ASYNC_LDS
            __builtin_amdgcn_global_load_async_to_lds_b128(
                (gv4i*)gsrc, (lv4i*)&Xs[r * 32 + ks], 0, 0);
#else
            *(v8bf*)&Xs[r * 32 + ks] = *(const v8bf*)gsrc;
#endif
        }
        {   // Wo fp32 rows -> bf16, K-contiguous per output column n
            int n = tid >> 2, ks = (tid & 3) * 8;
            const float4* src = (const float4*)(Wo + (size_t)(nt * 64 + n) * CD + kk + ks);
            if (kk + 32 < CD) __builtin_prefetch(Wo + (size_t)(nt * 64 + n) * CD + kk + 32 + ks, 0, 0);
            *(v8bf*)&Ws[n * 32 + ks] = pack8(src[0], src[1]);
        }
#if HAS_ASYNC_LDS
        asm volatile("s_wait_asynccnt 0x0" ::: "memory");
#endif
        __syncthreads();
        v16bf a  = frag_a(&Xs[rt * 16 * 32], 32, lane);
        v16bf b0 = frag_b(&Ws[(cg * 2 + 0) * 16 * 32], 32, lane);
        v16bf b1 = frag_b(&Ws[(cg * 2 + 1) * 16 * 32], 32, lane);
        c0 = wmma_bf16(a, b0, c0);
        c1 = wmma_bf16(a, b1, c1);
        __syncthreads();
    }

    const int col   = lane & 15;
    const int rbase = mt * 64 + rt * 16 + ((lane & 16) >> 1);
    const int n0 = nt * 64 + cg * 32 + col, n1 = n0 + 16;
    const float b0v = bo[n0], b1v = bo[n1];
    #pragma unroll
    for (int r = 0; r < 8; ++r) {
        int row = rbase + r;
        out[(size_t)row * CD + n0] = c0[r] + b0v;
        out[(size_t)row * CD + n1] = c1[r] + b1v;
    }
}

// ---------------------------------------------------------------------------
extern "C" void kernel_launch(void* const* d_in, const int* in_sizes, int n_in,
                              void* d_out, int out_size, void* d_ws, size_t ws_size,
                              hipStream_t stream)
{
    const float* x  = (const float*)d_in[0];
    const float* Wq = (const float*)d_in[1];
    const float* bq = (const float*)d_in[2];
    const float* Wk = (const float*)d_in[3];
    const float* bk = (const float*)d_in[4];
    const float* Wv = (const float*)d_in[5];
    const float* bv = (const float*)d_in[6];
    const float* Wo = (const float*)d_in[7];
    const float* bo = (const float*)d_in[8];

    float* out   = (float*)d_out;                       // [B,S,D]
    float* probs = out + (size_t)CB * CS * CD;          // [B,H,S,S]

    // workspace: q,k [B,H,S,E], v transposed [B,H,E,S], ctx [B,S,D]; all bf16
    const size_t qsz = (size_t)CB * CH * CS * CE;
    __bf16* qb  = (__bf16*)d_ws;
    __bf16* kb  = qb + qsz;
    __bf16* vb  = kb + qsz;
    __bf16* ctx = vb + qsz;

    qkv_proj_kernel<<<dim3(CS / 64, CB * CH, 3), 256, 0, stream>>>(
        x, Wq, bq, Wk, bk, Wv, bv, qb, kb, vb);
    scores_kernel<<<dim3(CS / 64, CS / 64, CB * CH), 256, 0, stream>>>(qb, kb, probs);
    softmax_kernel<<<dim3(CB * CH * CS / 8), 256, 0, stream>>>(probs);
    ctx_kernel<<<dim3(CS / 64, CB * CH), 256, 0, stream>>>(probs, vb, ctx);
    outproj_kernel<<<dim3(CB * CS / 64, CD / 64), 256, 0, stream>>>(ctx, Wo, bo, out);
}
